// ResidualAttentionBlock_22557168238690
// MI455X (gfx1250) — matrix operand order repaired and checked by reading, not compiled, not run on hardware
//
#include <hip/hip_runtime.h>
#include <stdint.h>

typedef __attribute__((ext_vector_type(16))) _Float16 v16h;
typedef __attribute__((ext_vector_type(8)))  float    v8f;
typedef __attribute__((ext_vector_type(4)))  int      v4i;

#if defined(__HIP_DEVICE_COMPILE__) && \
    __has_builtin(__builtin_amdgcn_global_load_async_to_lds_b128) && \
    __has_builtin(__builtin_amdgcn_s_wait_asynccnt)
#define USE_ASYNC_LDS 1
#else
#define USE_ASYNC_LDS 0
#endif

namespace {
constexpr int kB = 4;
constexpr int kT = 511;
constexpr int kD = 64;
constexpr int kH = 8;
constexpr int kTok = kB * kT;        // 2044
constexpr int kQCols = kH * kD;      // 512
constexpr int kVCols = 2 * kH * kD;  // 1024
constexpr float kScale = 0.125f;     // 1/sqrt(64)
constexpr float kHalfSun = 4.5f;     // SUN/2
}

// Element e of a v16h A-fragment (16x32 f16 A matrix, wave32):
// VGPR j=e/2, packed half p=e%2. K = (j<4?0:16) + (laneHi?8:0) + (j%4)*2 + p.
__device__ __forceinline__ int a_k_of(int e, int hi) {
  const int j = e >> 1, p = e & 1;
  const int grp = (j < 4) ? 0 : 16;
  const int jj = (j < 4) ? j : (j - 4);
  return grp + (hi ? 8 : 0) + (jj << 1) + p;
}

__global__ void zero_kernel(float* __restrict__ p, int n) {
  const int i = blockIdx.x * blockDim.x + threadIdx.x;
  if (i < n) p[i] = 0.0f;
}

// ---- K1: q/v projections via f16 WMMA, f32 accumulate -----------------------
__global__ __launch_bounds__(32) void proj_kernel(
    const float* __restrict__ x, const float* __restrict__ wq,
    const float* __restrict__ wv, float* __restrict__ Q,
    _Float16* __restrict__ V) {
  const int lane = threadIdx.x;
  const int hi = lane >> 4;
  const int m = lane & 15;
  const int ttile = blockIdx.x;   // 128 token tiles (2044 tokens, clamped)
  const int ntile = blockIdx.y;   // 96 output tiles: 32 for q, 64 for v
  const int n = ntile * 16 + m;   // B-fragment / D-fragment column

  int tokA = ttile * 16 + m;      // A-fragment row (lane = M)
  if (tokA > kTok - 1) tokA = kTok - 1;

  v8f acc = {};
  #pragma unroll
  for (int kc = 0; kc < kD; kc += 32) {
    v16h a, bf;
    #pragma unroll
    for (int e = 0; e < 16; ++e)
      a[e] = (_Float16)x[tokA * kD + kc + a_k_of(e, hi)];
    #pragma unroll
    for (int e = 0; e < 16; ++e) {   // B layout: lane = N, K = a_k_of(e, hi)
      const int k = kc + a_k_of(e, hi);
      const float w = (n < kQCols) ? wq[n * (kD + 1) + k]
                                   : wv[(n - kQCols) * (kD + 1) + k];
      bf[e] = (_Float16)w;
    }
    acc = __builtin_amdgcn_wmma_f32_16x16x32_f16(false, a, false, bf,
                                                 (short)0, acc, false, false);
  }
  const float bias = (n < kQCols) ? wq[n * (kD + 1) + kD]
                                  : wv[(n - kQCols) * (kD + 1) + kD];
  #pragma unroll
  for (int r = 0; r < 8; ++r) {     // D layout: M = r + hi*8, N = lane&15
    const int tok = ttile * 16 + r + hi * 8;
    if (tok < kTok) {
      const float val = acc[r] + bias;
      if (n < kQCols) Q[tok * kQCols + n] = val;
      else            V[tok * kVCols + (n - kQCols)] = (_Float16)val;
    }
  }
}

// ---- K2: softmax normalizers Z[b,h,t] = 1 + sum_s exp(sc[s,t]) --------------
__global__ __launch_bounds__(128) void zsum_kernel(
    const float* __restrict__ x, const float* __restrict__ Q,
    const float* __restrict__ wk, float* __restrict__ Z) {
  const int t = blockIdx.x, h = blockIdx.y, b = blockIdx.z;
  __shared__ float qs[kD], wks[kD];
  __shared__ float red[128];
  const int tid = threadIdx.x;
  if (tid < kD) {
    qs[tid]  = Q[(b * kT + t) * kQCols + h * kD + tid];
    wks[tid] = wk[h * kD + tid];
  }
  __syncthreads();
  float part = 0.0f;
  for (int s = tid; s < kT; s += 128) {
    const float* xs = x + (b * kT + s) * kD;
    float l1 = 0.0f;
    #pragma unroll 8
    for (int d = 0; d < kD; ++d) l1 += fabsf(qs[d] - xs[d] * wks[d]);
    part += __expf(-kScale * l1);
  }
  red[tid] = part;
  __syncthreads();
  #pragma unroll
  for (int off = 64; off > 0; off >>= 1) {
    if (tid < off) red[tid] += red[tid + off];
    __syncthreads();
  }
  if (tid == 0) Z[(b * kH + h) * kT + t] = red[0] + 1.0f;  // +1 = sink
}

// ---- K3: fused attention: recompute E tiles, WMMA (A1*Vf + A2*Vb) -----------
__global__ __launch_bounds__(32) void attn_kernel(
    const float* __restrict__ x, const float* __restrict__ Q,
    const _Float16* __restrict__ V, const float* __restrict__ wk,
    const float* __restrict__ Z, float* __restrict__ bsum) {
  const int lane = threadIdx.x;
  const int hi = lane >> 4;
  const int m = lane & 15;
  const int ttile = blockIdx.x, h = blockIdx.y, b = blockIdx.z;

  // Row stride 68 floats = 272B: 16B-aligned rows (async b128 OK) and
  // bank-skewed (68 mod 64 = 4) so 16 distinct rows never collide.
  __shared__ __align__(16) float kch[32][68];  // k rows of current chunk
  __shared__ __align__(16) float qch[32][68];  // q rows of current chunk
  __shared__ __align__(16) float qts[16][68];  // q rows of the 16 dest tokens
  __shared__ __align__(16) float kts[16][68];  // k rows of the 16 dest tokens

  {
    int t = ttile * 16 + m;
    if (t > kT - 1) t = kT - 1;
    for (int d = hi * 32; d < hi * 32 + 32; ++d) {
      const float xv = x[(b * kT + t) * kD + d];
      qts[m][d] = Q[(b * kT + t) * kQCols + h * kD + d];
      kts[m][d] = xv * wk[h * kD + d];
    }
  }

  int tmine = ttile * 16 + m;
  if (tmine > kT - 1) tmine = kT - 1;
  const float inv_zt = 1.0f / Z[(b * kH + h) * kT + tmine];

  int sl[16];
  #pragma unroll
  for (int e = 0; e < 16; ++e) sl[e] = a_k_of(e, hi);

  v8f acc[4] = {};

  for (int ch = 0; ch < 16; ++ch) {
    __syncthreads();

    // Prefetch next chunk's source lines while we chew on this one.
    if (ch + 1 < 16) {
      int sn = (ch + 1) * 32 + lane;
      if (sn > kT - 1) sn = kT - 1;
      __builtin_prefetch(x + (size_t)(b * kT + sn) * kD, 0, 3);
      __builtin_prefetch(Q + (size_t)(b * kT + sn) * kQCols + h * kD, 0, 3);
      __builtin_prefetch(V + (size_t)(b * kT + sn) * kVCols + h * kD, 0, 3);
    }

#if USE_ASYNC_LDS
    // Stage the 32 q rows of this chunk straight into LDS with async DMA:
    // lane-job (i, lane) copies 16B; row r = 2i+hi, 16B-slice p = lane&15.
    #pragma unroll
    for (int i = 0; i < 16; ++i) {
      const int r = i * 2 + hi;
      int s = ch * 32 + r;
      if (s > kT - 1) s = kT - 1;
      const float* gsrc = Q + (size_t)(b * kT + s) * kQCols + h * kD + m * 4;
      __builtin_amdgcn_global_load_async_to_lds_b128(
          (v4i*)(uintptr_t)gsrc, (v4i*)&qch[r][m * 4], 0, 0);
    }
#endif
    // k rows need the x*wk transform, so they go through VGPRs.
    {
      int s = ch * 32 + lane;
      if (s > kT - 1) s = kT - 1;
      for (int d = 0; d < kD; ++d) {
        const float xv = x[(b * kT + s) * kD + d];
        kch[lane][d] = xv * wk[h * kD + d];
#if !USE_ASYNC_LDS
        qch[lane][d] = Q[(b * kT + s) * kQCols + h * kD + d];
#endif
      }
    }
#if USE_ASYNC_LDS
    __builtin_amdgcn_s_wait_asynccnt(0);
#endif
    __syncthreads();

    float s1[16] = {0};
    float s2[16] = {0};
    #pragma unroll 1
    for (int d = 0; d < kD; ++d) {
      const float qv = qts[m][d];
      const float kv = kts[m][d];
      #pragma unroll
      for (int e = 0; e < 16; ++e) {
        s1[e] += fabsf(qv - kch[sl[e]][d]);   // |q[t] - k[s]|  (forward)
        s2[e] += fabsf(qch[sl[e]][d] - kv);   // |q[s] - k[t]|  (backward)
      }
    }

    v16h a1, a2;
    #pragma unroll
    for (int e = 0; e < 16; ++e) {
      const int s = ch * 32 + sl[e];
      float e1 = 0.0f, e2 = 0.0f;
      if (s < kT) {
        e1 = __expf(-kScale * s1[e]) * inv_zt;                      // E[s,t]/Z[t]
        e2 = __expf(-kScale * s2[e]) / Z[(b * kH + h) * kT + s];    // E[t,s]/Z[s]
      }
      a1[e] = (_Float16)e1;
      a2[e] = (_Float16)e2;
    }

    #pragma unroll
    for (int wt = 0; wt < 4; ++wt) {
      v16h bfv, bbv;
      #pragma unroll
      for (int e = 0; e < 16; ++e) {   // B layout: lane = N, K = a_k_of(e, hi)
        int s = ch * 32 + sl[e];
        if (s > kT - 1) s = kT - 1;    // padded rows nulled via A operand
        const _Float16* vp = V + (size_t)(b * kT + s) * kVCols;
        bfv[e] = vp[h * kD + wt * 16 + m];
        bbv[e] = vp[kQCols + h * kD + wt * 16 + m];
      }
      acc[wt] = __builtin_amdgcn_wmma_f32_16x16x32_f16(
          false, a1, false, bfv, (short)0, acc[wt], false, false);
      acc[wt] = __builtin_amdgcn_wmma_f32_16x16x32_f16(
          false, a2, false, bbv, (short)0, acc[wt], false, false);
    }
  }

  #pragma unroll
  for (int wt = 0; wt < 4; ++wt) {
    #pragma unroll
    for (int r = 0; r < 8; ++r) {
      const int t = ttile * 16 + r + hi * 8;
      if (t < kT)
        atomicAdd(&bsum[(b * kT + t) * kD + wt * 16 + m], acc[wt][r]);
    }
  }
}

// ---- K4: shifted quickGELU + 64x64 fanout GEMM (WMMA) + residual ------------
__global__ __launch_bounds__(32) void fanout_kernel(
    const float* __restrict__ x, const float* __restrict__ bsum,
    const float* __restrict__ fw, float* __restrict__ out) {
  const int lane = threadIdx.x;
  const int hi = lane >> 4;
  const int m = lane & 15;
  const int ttile = blockIdx.x;
  const int ntile = blockIdx.y;
  const int n = ntile * 16 + m;

  int tokA = ttile * 16 + m;
  if (tokA > kTok - 1) tokA = kTok - 1;

  v8f acc = {};
  #pragma unroll
  for (int kc = 0; kc < kD; kc += 32) {
    v16h a, bf;
    #pragma unroll
    for (int e = 0; e < 16; ++e) {
      const float z = bsum[tokA * kD + kc + a_k_of(e, hi)] + kHalfSun;
      const float g = z / (1.0f + __expf(-1.702f * z)) - kHalfSun;  // quickGELU
      a[e] = (_Float16)g;
    }
    #pragma unroll
    for (int e = 0; e < 16; ++e)
      bf[e] = (_Float16)fw[n * (kD + 1) + kc + a_k_of(e, hi)];
    acc = __builtin_amdgcn_wmma_f32_16x16x32_f16(false, a, false, bf,
                                                 (short)0, acc, false, false);
  }
  const float bias = fw[n * (kD + 1) + kD];
  #pragma unroll
  for (int r = 0; r < 8; ++r) {
    const int tok = ttile * 16 + r + hi * 8;
    if (tok < kTok) out[tok * kD + n] = x[tok * kD + n] + acc[r] + bias;
  }
}

extern "C" void kernel_launch(void* const* d_in, const int* in_sizes, int n_in,
                              void* d_out, int out_size, void* d_ws,
                              size_t ws_size, hipStream_t stream) {
  (void)in_sizes; (void)n_in; (void)out_size; (void)ws_size;
  const float* x  = (const float*)d_in[0];
  const float* wq = (const float*)d_in[1];
  const float* wv = (const float*)d_in[2];
  const float* wk = (const float*)d_in[3];
  const float* fw = (const float*)d_in[4];
  float* out = (float*)d_out;

  // Workspace layout (~9 MB, fits L2 with room to spare):
  char* ws = (char*)d_ws;
  float*    Q    = (float*)ws;                                  // 2044*512 f32
  _Float16* V    = (_Float16*)(ws + (size_t)kTok * kQCols * 4); // 2044*1024 f16
  float*    Z    = (float*)(ws + (size_t)kTok * kQCols * 4 +
                                 (size_t)kTok * kVCols * 2);    // 4*8*511 f32
  float*    bsum = Z + kB * kH * kT;                            // 2044*64 f32

  const int nb = kTok * kD;
  zero_kernel<<<(nb + 255) / 256, 256, 0, stream>>>(bsum, nb);
  proj_kernel<<<dim3(128, 96), 32, 0, stream>>>(x, wq, wv, Q, V);
  zsum_kernel<<<dim3(kT, kH, kB), 128, 0, stream>>>(x, Q, wk, Z);
  attn_kernel<<<dim3(32, kH, kB), 32, 0, stream>>>(x, Q, V, wk, Z, bsum);
  fanout_kernel<<<dim3(128, 4), 32, 0, stream>>>(x, bsum, fw, out);
}
